// GNN_codebook_51110110822776
// MI455X (gfx1250) — matrix operand order repaired and checked by reading, not compiled, not run on hardware
//
#include <hip/hip_runtime.h>

#define D   128
#define D2  256
#define NUM_BOND_TYPE 7
#define NUM_BOND_DIR  3
#define SELF_LOOP_BOND 4

typedef __attribute__((ext_vector_type(16))) __bf16 v16bf;
typedef __attribute__((ext_vector_type(8)))  float  v8f;

// h[n] = x_emb1[attr0] + x_emb2[attr1]   (one wave = one node row, float4/lane)
__global__ void k_node_embed(const int* __restrict__ node_attr,
                             const float* __restrict__ e1,
                             const float* __restrict__ e2,
                             float* __restrict__ h, int n) {
    long long t = (long long)blockIdx.x * blockDim.x + threadIdx.x;
    if (t >= (long long)n * 32) return;
    int node = (int)(t >> 5);
    int c    = (int)(t & 31) * 4;
    int a0 = node_attr[2 * node];
    int a1 = node_attr[2 * node + 1];
    float4 v1 = *(const float4*)(e1 + (long long)a0 * D + c);
    float4 v2 = *(const float4*)(e2 + (long long)a1 * D + c);
    float4 o; o.x = v1.x + v2.x; o.y = v1.y + v2.y; o.z = v1.z + v2.z; o.w = v1.w + v2.w;
    *(float4*)(h + (long long)node * D + c) = o;
}

// agg[n] = h[n] + ee1l[SELF_LOOP_BOND] + ee2l[0]   (self-loop absorbed into init)
__global__ void k_selfloop(const float* __restrict__ h,
                           const float* __restrict__ ee1l,
                           const float* __restrict__ ee2l,
                           float* __restrict__ agg, int n) {
    long long t = (long long)blockIdx.x * blockDim.x + threadIdx.x;
    if (t >= (long long)n * 32) return;
    int node = (int)(t >> 5);
    int c    = (int)(t & 31) * 4;
    float4 hv = *(const float4*)(h + (long long)node * D + c);
    float4 b1 = *(const float4*)(ee1l + SELF_LOOP_BOND * D + c);
    float4 b2 = *(const float4*)(ee2l + 0 * D + c);
    float4 o; o.x = hv.x + b1.x + b2.x; o.y = hv.y + b1.y + b2.y;
    o.z = hv.z + b1.z + b2.z; o.w = hv.w + b1.w + b2.w;
    *(float4*)(agg + (long long)node * D + c) = o;
}

// one wave per edge: gather h[src], add edge embedding, hw-atomic scatter to agg[dst]
__global__ void k_edges(const float* __restrict__ h,
                        const int* __restrict__ esrc,
                        const int* __restrict__ edst,
                        const int* __restrict__ eattr,
                        const float* __restrict__ ee1l,
                        const float* __restrict__ ee2l,
                        float* __restrict__ agg, long long E) {
    long long t = (long long)blockIdx.x * blockDim.x + threadIdx.x;
    if (t >= E * 32) return;
    long long e = t >> 5;
    int c = (int)(t & 31) * 4;
    int src = esrc[e];
    int dst = edst[e];
    int a0  = eattr[2 * e];
    int a1  = eattr[2 * e + 1];
    float4 hv = *(const float4*)(h + (long long)src * D + c);
    float4 b1 = *(const float4*)(ee1l + (long long)a0 * D + c);
    float4 b2 = *(const float4*)(ee2l + (long long)a1 * D + c);
    float* p = agg + (long long)dst * D + c;
    unsafeAtomicAdd(p + 0, hv.x + b1.x + b2.x);
    unsafeAtomicAdd(p + 1, hv.y + b1.y + b2.y);
    unsafeAtomicAdd(p + 2, hv.z + b1.z + b2.z);
    unsafeAtomicAdd(p + 3, hv.w + b1.w + b2.w);
}

// Pack W[K,NCOLS] f32 -> bf16 B-fragment order: Wp[((ct*KSTEPS + kk)*32 + lane)*16 + pos]
// B fragment order used here (contiguous k, same as A staging): lane = n + 16*(kin>>4), pos = kin & 15.
__global__ void k_packW(const float* __restrict__ W, __bf16* __restrict__ Wp,
                        int K, int NCOLS) {
    int idx = blockIdx.x * blockDim.x + threadIdx.x;
    if (idx >= K * NCOLS) return;
    int k = idx / NCOLS, cidx = idx % NCOLS;
    int ct = cidx >> 4, nn = cidx & 15;
    int kk = k >> 5,    kin = k & 31;
    int lane = nn + ((kin >> 4) << 4);
    int pos  = kin & 15;
    int ksteps = K >> 5;
    Wp[(((size_t)ct * ksteps + kk) * 32 + lane) * 16 + pos] = (__bf16)W[idx];
}

// Out[N,NCOLS] = act(A[N,K] @ W[K,NCOLS] + bias)
// Block: 16-row slab of A staged in LDS in WMMA fragment order, then preloaded
// once into registers per wave. Wave: NCOLS/128 16x16 output tiles; fully
// unrolled v_wmma_f32_16x16x32_bf16 chain; B fragment = one 32B global load.
template <int K, int NCOLS>
__global__ void k_gemm(const float* __restrict__ A,
                       const __bf16* __restrict__ Wp,
                       const float* __restrict__ bias,
                       float* __restrict__ Out, int n, int do_relu) {
    constexpr int KSTEPS = K / 32;
    constexpr int REPS   = NCOLS / 128;   // col tiles per wave (8 waves x 16 cols)
    __shared__ alignas(32) __bf16 Af[KSTEPS * 32 * 16];  // [kk][lane][16]

    const int tid = threadIdx.x;
    const int rowBase = blockIdx.x * 16;

    // Cooperative stage of A slab into fragment order.
    // A fragment order used here (contiguous k, same as packed B): lane = m + 16*(kin>>4), pos = kin & 15.
    for (int i = tid; i < 16 * K; i += 256) {
        int r = i / K, c = i % K;
        int gr = rowBase + r;
        float v = (gr < n) ? A[(long long)gr * K + c] : 0.0f;
        int kk  = c >> 5;
        int kin = c & 31;
        int lane = r + ((kin >> 4) << 4);
        int pos  = kin & 15;
        Af[(kk * 32 + lane) * 16 + pos] = (__bf16)v;
    }
    __syncthreads();

    const int wave = tid >> 5;
    const int ln   = tid & 31;
    const int m16  = ln & 15;           // output column within tile
    const int mOff = (ln >> 4) << 3;    // output row offset within tile

    // Preload all A fragments once (reused across col tiles).
    v16bf afrag[KSTEPS];
#pragma unroll
    for (int kk = 0; kk < KSTEPS; ++kk)
        afrag[kk] = *(const v16bf*)(Af + (kk * 32 + ln) * 16);

#pragma unroll
    for (int rep = 0; rep < REPS; ++rep) {
        const int ct = wave + rep * 8;
        v8f acc = {0.f, 0.f, 0.f, 0.f, 0.f, 0.f, 0.f, 0.f};
#pragma unroll
        for (int kk = 0; kk < KSTEPS; ++kk) {
            v16bf b = *(const v16bf*)(Wp + (((size_t)ct * KSTEPS + kk) * 32 + ln) * 16);
            acc = __builtin_amdgcn_wmma_f32_16x16x32_bf16(
                false, afrag[kk], false, b, (short)0, acc, false, false);
        }
        const int colBase = ct * 16;
        const float bv = bias[colBase + m16];
        if (rowBase + 15 < n) {
            // Whole-tile fast path: one base address, 8 stores at immediate offsets.
            float* p = Out + (size_t)(rowBase + mOff) * NCOLS + colBase + m16;
#pragma unroll
            for (int r = 0; r < 8; ++r) {
                float val = acc[r] + bv;
                val = do_relu ? fmaxf(val, 0.0f) : val;
                p[r * NCOLS] = val;
            }
        } else {
#pragma unroll
            for (int r = 0; r < 8; ++r) {
                int grow = rowBase + mOff + r;
                if (grow < n) {
                    float val = acc[r] + bv;
                    val = do_relu ? fmaxf(val, 0.0f) : val;
                    Out[(size_t)grow * NCOLS + colBase + m16] = val;
                }
            }
        }
    }
}

extern "C" void kernel_launch(void* const* d_in, const int* in_sizes, int n_in,
                              void* d_out, int out_size, void* d_ws, size_t ws_size,
                              hipStream_t stream) {
    const int*   node_attr  = (const int*)d_in[0];
    const int*   edge_index = (const int*)d_in[1];
    const int*   edge_attr  = (const int*)d_in[2];
    const float* x_emb1     = (const float*)d_in[3];
    const float* x_emb2     = (const float*)d_in[4];
    const float* ee1        = (const float*)d_in[5];
    const float* ee2        = (const float*)d_in[6];
    const float* W1         = (const float*)d_in[7];
    const float* b1         = (const float*)d_in[8];
    const float* W2         = (const float*)d_in[9];
    const float* b2         = (const float*)d_in[10];

    const int       n = in_sizes[0] / 2;
    const long long E = in_sizes[1] / 2;
    const int       L = in_sizes[7] / (D * D2);

    float*  bufA = (float*)d_ws;                    // h ping
    float*  bufB = bufA + (size_t)n * D;            // agg / h pong
    float*  tmp  = bufB + (size_t)n * D;            // [N, 2D] hidden
    __bf16* Wp1  = (__bf16*)(tmp + (size_t)n * D2); // packed W1 fragments (64KB)
    __bf16* Wp2  = Wp1 + (size_t)D * D2;            // packed W2 fragments (64KB)
    float*  out  = (float*)d_out;

    const long long tN   = (long long)n * 32;
    const int       blkN = (int)((tN + 255) / 256);
    const long long tE   = E * 32;
    const int       blkE = (int)((tE + 255) / 256);
    const int       gblk = (n + 15) / 16;
    const int       pblk = (D * D2 + 255) / 256;

    k_node_embed<<<blkN, 256, 0, stream>>>(node_attr, x_emb1, x_emb2, bufA, n);

    float* h_in  = bufA;
    float* h_agg = bufB;
    for (int l = 0; l < L; ++l) {
        const float* ee1l = ee1 + (size_t)l * NUM_BOND_TYPE * D;
        const float* ee2l = ee2 + (size_t)l * NUM_BOND_DIR * D;

        k_packW<<<pblk, 256, 0, stream>>>(W1 + (size_t)l * D * D2, Wp1, D, D2);
        k_packW<<<pblk, 256, 0, stream>>>(W2 + (size_t)l * D2 * D, Wp2, D2, D);

        k_selfloop<<<blkN, 256, 0, stream>>>(h_in, ee1l, ee2l, h_agg, n);
        k_edges<<<blkE, 256, 0, stream>>>(h_in, edge_index, edge_index + E, edge_attr,
                                          ee1l, ee2l, h_agg, E);

        k_gemm<D, D2><<<gblk, 256, 0, stream>>>(
            h_agg, Wp1, b1 + (size_t)l * D2, tmp, n, 1);

        float* h_out = (l == L - 1) ? out : h_agg;  // gemm2 only reads tmp: safe to overwrite agg
        k_gemm<D2, D><<<gblk, 256, 0, stream>>>(
            tmp, Wp2, b2 + (size_t)l * D, h_out, n, (l != L - 1) ? 1 : 0);

        float* t2 = h_in;
        h_in  = h_out;
        h_agg = t2;
    }
}
